// RotarySelfAttention_6107443494996
// MI455X (gfx1250) — matrix operand order repaired and checked by reading, not compiled, not run on hardware
//
#include <hip/hip_runtime.h>
#include <hip/hip_bf16.h>

typedef __attribute__((ext_vector_type(16))) _Float16 v16h;
typedef __attribute__((ext_vector_type(8)))  _Float16 v8h;
typedef __attribute__((ext_vector_type(8)))  float    v8f;

union Frag { v16h v; v8h h[2]; };

__device__ __forceinline__ v8f vzero8() {
  v8f z;
#pragma unroll
  for (int i = 0; i < 8; ++i) z[i] = 0.0f;
  return z;
}

#define HEADS 8
#define DH    64
#define SEQ   4096
#define BATCH 2
#define DIM   512
#define INNER 512
#define QKV_N 1536
#define ROWS  (BATCH * SEQ) /* 8192 */

// ---------------------------------------------------------------------------
// 1) LayerNorm: x[row][512] f32 -> xn f16
// ---------------------------------------------------------------------------
__global__ __launch_bounds__(256)
void k_ln(const float* __restrict__ x, const float* __restrict__ w,
          const float* __restrict__ b, _Float16* __restrict__ xn) {
  int row = blockIdx.x;
  int t = threadIdx.x;
  const float* xr = x + (size_t)row * DIM;
  float v0 = xr[t], v1 = xr[t + 256];
  __shared__ float s_sum[256], s_sq[256];
  s_sum[t] = v0 + v1;
  s_sq[t]  = v0 * v0 + v1 * v1;
  __syncthreads();
  for (int off = 128; off > 0; off >>= 1) {
    if (t < off) { s_sum[t] += s_sum[t + off]; s_sq[t] += s_sq[t + off]; }
    __syncthreads();
  }
  __shared__ float mu_s, rs_s;
  if (t == 0) {
    float mu  = s_sum[0] * (1.0f / DIM);
    float var = s_sq[0] * (1.0f / DIM) - mu * mu;
    mu_s = mu;
    rs_s = rsqrtf(var + 1e-5f);
  }
  __syncthreads();
  float mu = mu_s, rs = rs_s;
  xn[(size_t)row * DIM + t]       = (_Float16)((v0 - mu) * rs * w[t] + b[t]);
  xn[(size_t)row * DIM + t + 256] = (_Float16)((v1 - mu) * rs * w[t + 256] + b[t + 256]);
}

// ---------------------------------------------------------------------------
// f32 -> f16 convert (weights)
// ---------------------------------------------------------------------------
__global__ void k_cvt(const float* __restrict__ s, _Float16* __restrict__ d, int n) {
  int i = blockIdx.x * 256 + threadIdx.x;
  if (i < n) d[i] = (_Float16)s[i];
}

// ---------------------------------------------------------------------------
// 2)/6) WMMA GEMM: C[M][N] = A[M][K] * W[N][K]^T
//   block tile 32(M) x 256(N), 8 waves (2x4), wave tile 16x64, K-step 32.
//   outH!=null : write f16 ; else write f32 + bias
// ---------------------------------------------------------------------------
__global__ __launch_bounds__(256)
void k_gemm(const _Float16* __restrict__ A, const _Float16* __restrict__ W,
            int K, _Float16* __restrict__ outH, float* __restrict__ outF,
            const float* __restrict__ bias, int N) {
  __shared__ _Float16 At[32 * 32];
  __shared__ _Float16 Bt[256 * 32];
  int t = threadIdx.x;
  int lane = t & 31, wid = t >> 5;
  int waveM = wid >> 2, waveN = wid & 3;
  int m0 = blockIdx.x * 32;
  int n0 = blockIdx.y * 256;
  int lrow = lane & 15;
  int kb8  = (lane >> 4) * 8;   // A-frag K base within 16-half group
  int k16  = (lane >> 4) * 16;  // unused; B-frag also uses kb8

  v8f acc[4];
#pragma unroll
  for (int j = 0; j < 4; ++j) acc[j] = vzero8();

  int arow = lrow + waveM * 16;

  for (int k0 = 0; k0 < K; k0 += 32) {
    if (t < 128) {
      int r = t >> 2, c = (t & 3) * 8;
      *(v8h*)&At[r * 32 + c] = *(const v8h*)&A[(size_t)(m0 + r) * K + k0 + c];
    }
#pragma unroll
    for (int i = 0; i < 4; ++i)
      *(v8h*)&Bt[t * 32 + i * 8] = *(const v8h*)&W[(size_t)(n0 + t) * K + k0 + i * 8];
    // prefetch next K-chunk while this one is consumed
    if (k0 + 32 < K) {
      __builtin_prefetch(&W[(size_t)(n0 + t) * K + k0 + 32], 0, 0);
      if (t < 128) __builtin_prefetch(&A[(size_t)(m0 + (t >> 2)) * K + k0 + 32], 0, 0);
    }
    __syncthreads();

    Frag a;
    a.h[0] = *(const v8h*)&At[arow * 32 + kb8];
    a.h[1] = *(const v8h*)&At[arow * 32 + 16 + kb8];
    Frag bf[4];
#pragma unroll
    for (int j = 0; j < 4; ++j) {
      int nl = waveN * 64 + j * 16 + lrow;
      bf[j].h[0] = *(const v8h*)&Bt[nl * 32 + kb8];
      bf[j].h[1] = *(const v8h*)&Bt[nl * 32 + 16 + kb8];
    }
#pragma unroll
    for (int j = 0; j < 4; ++j)
      acc[j] = __builtin_amdgcn_wmma_f32_16x16x32_f16(false, a.v, false, bf[j].v,
                                                      (short)0, acc[j], false, false);
    __syncthreads();
  }

  int rbase = (lane >> 4) * 8;
#pragma unroll
  for (int j = 0; j < 4; ++j) {
    int col = n0 + waveN * 64 + j * 16 + lrow;
#pragma unroll
    for (int r = 0; r < 8; ++r) {
      int m = m0 + waveM * 16 + r + rbase;
      float v = acc[j][r];
      if (outH) outH[(size_t)m * N + col] = (_Float16)v;
      else      outF[(size_t)m * N + col] = v + bias[col];
    }
  }
}

// ---------------------------------------------------------------------------
// 3) RoPE + scatter to head layout. q scaled by 0.125 (= dim_head^-0.5).
//    v stored TRANSPOSED: vT[bh][d][n]
// ---------------------------------------------------------------------------
__global__ __launch_bounds__(256)
void k_rope_scatter(const _Float16* __restrict__ qkv, const float* __restrict__ rot,
                    _Float16* __restrict__ qh, _Float16* __restrict__ kh,
                    _Float16* __restrict__ vT) {
  int m = blockIdx.x;          // b*4096 + n
  int b = m >> 12, n = m & 4095;
  int t = threadIdx.x;
  int h = t >> 5, pr = t & 31;
  int d0 = pr * 2, d1 = d0 + 1;
  const float* f = rot + (size_t)m * DH;
  float f0 = f[d0], f1 = f[d1];
  float c0 = __cosf(f0), s0 = __sinf(f0);
  float c1 = __cosf(f1), s1 = __sinf(f1);
  int bh = b * HEADS + h;
  const _Float16* base = qkv + (size_t)m * QKV_N + h * DH;
  size_t oh = ((size_t)bh * SEQ + n) * DH;
  {
    float x0 = (float)base[d0], x1 = (float)base[d1];
    float y0 = x0 * c0 - x1 * s0;
    float y1 = x1 * c1 + x0 * s1;
    qh[oh + d0] = (_Float16)(y0 * 0.125f);
    qh[oh + d1] = (_Float16)(y1 * 0.125f);
  }
  {
    float x0 = (float)base[INNER + d0], x1 = (float)base[INNER + d1];
    float y0 = x0 * c0 - x1 * s0;
    float y1 = x1 * c1 + x0 * s1;
    kh[oh + d0] = (_Float16)y0;
    kh[oh + d1] = (_Float16)y1;
  }
  {
    float x0 = (float)base[2 * INNER + d0], x1 = (float)base[2 * INNER + d1];
    float y0 = x0 * c0 - x1 * s0;
    float y1 = x1 * c1 + x0 * s1;
    vT[((size_t)bh * DH + d0) * SEQ + n] = (_Float16)y0;
    vT[((size_t)bh * DH + d1) * SEQ + n] = (_Float16)y1;
  }
}

// ---------------------------------------------------------------------------
// 4) Flash attention. Grid (SEQ/128, B*H). 8 waves, 16 q-rows each.
//    Key tiles of 32; online softmax; all matmuls on v_wmma f16.
// ---------------------------------------------------------------------------
__global__ __launch_bounds__(256)
void k_attn(const _Float16* __restrict__ qh, const _Float16* __restrict__ kh,
            const _Float16* __restrict__ vT, _Float16* __restrict__ O) {
  __shared__ _Float16 Kt[32 * 64];      // [key][d]
  __shared__ _Float16 Vt[64 * 32];      // [d][key]   (from vT)
  __shared__ _Float16 Pt[8][16 * 32];   // per-wave P staging

  int t = threadIdx.x, lane = t & 31, wid = t >> 5;
  int bh = blockIdx.y;
  int qbase = blockIdx.x * 128 + wid * 16;
  int lrow = lane & 15, lhi = lane >> 4;
  int kb8 = lhi * 8, k16 = lhi * 16;

  Frag qa[2];
  const _Float16* qp = qh + ((size_t)bh * SEQ + qbase + lrow) * DH;
#pragma unroll
  for (int c = 0; c < 2; ++c) {
    qa[c].h[0] = *(const v8h*)(qp + c * 32 + kb8);
    qa[c].h[1] = *(const v8h*)(qp + c * 32 + 16 + kb8);
  }

  v8f acc[4];
#pragma unroll
  for (int j = 0; j < 4; ++j) acc[j] = vzero8();
  float mrun[8], lrun[8];
#pragma unroll
  for (int r = 0; r < 8; ++r) { mrun[r] = -1e30f; lrun[r] = 0.0f; }

  const _Float16* kg = kh + (size_t)bh * SEQ * DH;
  const _Float16* vg = vT + (size_t)bh * DH * SEQ;

  int kr = t >> 3, kc = (t & 7) * 8;    // K tile cooperative-load coords
  int vd = t >> 2, vc = (t & 3) * 8;    // V tile cooperative-load coords

  for (int kt = 0; kt < SEQ / 32; ++kt) {
    __syncthreads();   // previous iteration done with Kt/Vt
    *(v8h*)&Kt[kr * 64 + kc] = *(const v8h*)&kg[(size_t)(kt * 32 + kr) * DH + kc];
    *(v8h*)&Vt[vd * 32 + vc] = *(const v8h*)&vg[(size_t)vd * SEQ + kt * 32 + vc];
    // prefetch next tile (speculative; dropped if past end)
    if (kt + 1 < SEQ / 32) {
      __builtin_prefetch(&kg[(size_t)(kt * 32 + 32 + kr) * DH + kc], 0, 0);
      __builtin_prefetch(&vg[(size_t)vd * SEQ + kt * 32 + 32 + vc], 0, 0);
    }
    __syncthreads();

    // S = Q * K^T   (16 x 32, two 16-col blocks) -- batch all K frags first
    Frag kf[2][2];
#pragma unroll
    for (int j = 0; j < 2; ++j) {
      int key = j * 16 + lrow;
#pragma unroll
      for (int c = 0; c < 2; ++c) {
        kf[j][c].h[0] = *(const v8h*)&Kt[key * 64 + c * 32 + kb8];
        kf[j][c].h[1] = *(const v8h*)&Kt[key * 64 + c * 32 + 16 + kb8];
      }
    }
    v8f s[2];
#pragma unroll
    for (int j = 0; j < 2; ++j) {
      s[j] = vzero8();
#pragma unroll
      for (int c = 0; c < 2; ++c)
        s[j] = __builtin_amdgcn_wmma_f32_16x16x32_f16(false, qa[c].v, false, kf[j][c].v,
                                                      (short)0, s[j], false, false);
    }

    // online softmax (scale folded into q)
    float corr[8];
#pragma unroll
    for (int r = 0; r < 8; ++r) {
      float mx = fmaxf(s[0][r], s[1][r]);
#pragma unroll
      for (int off = 1; off < 16; off <<= 1) mx = fmaxf(mx, __shfl_xor(mx, off, 32));
      float mn = fmaxf(mrun[r], mx);
      corr[r] = __expf(mrun[r] - mn);
      float p0 = __expf(s[0][r] - mn);
      float p1 = __expf(s[1][r] - mn);
      float rs = p0 + p1;
#pragma unroll
      for (int off = 1; off < 16; off <<= 1) rs += __shfl_xor(rs, off, 32);
      lrun[r] = lrun[r] * corr[r] + rs;
      mrun[r] = mn;
      s[0][r] = p0;
      s[1][r] = p1;
    }
#pragma unroll
    for (int j = 0; j < 4; ++j)
#pragma unroll
      for (int r = 0; r < 8; ++r) acc[j][r] *= corr[r];

    // P: C-layout -> LDS -> A-layout fragment
    _Float16* P = Pt[wid];
#pragma unroll
    for (int r = 0; r < 8; ++r) {
      int rr = r + lhi * 8;
      P[rr * 32 + lrow]      = (_Float16)s[0][r];
      P[rr * 32 + 16 + lrow] = (_Float16)s[1][r];
    }
    __syncthreads();

    Frag pa;
    pa.h[0] = *(const v8h*)&P[lrow * 32 + kb8];
    pa.h[1] = *(const v8h*)&P[lrow * 32 + 16 + kb8];
    Frag vf[4];
#pragma unroll
    for (int jd = 0; jd < 4; ++jd) {
      int d = jd * 16 + lrow;
      vf[jd].h[0] = *(const v8h*)&Vt[d * 32 + kb8];
      vf[jd].h[1] = *(const v8h*)&Vt[d * 32 + 16 + kb8];
    }
#pragma unroll
    for (int jd = 0; jd < 4; ++jd)
      acc[jd] = __builtin_amdgcn_wmma_f32_16x16x32_f16(false, pa.v, false, vf[jd].v,
                                                       (short)0, acc[jd], false, false);
  }

  // normalize + store O[bh][n][d] f16
#pragma unroll
  for (int jd = 0; jd < 4; ++jd) {
    int d = jd * 16 + lrow;
#pragma unroll
    for (int r = 0; r < 8; ++r) {
      int row = qbase + r + lhi * 8;
      O[((size_t)bh * SEQ + row) * DH + d] = (_Float16)(acc[jd][r] / lrun[r]);
    }
  }
}

// ---------------------------------------------------------------------------
// 5) inverse RoPE + gather heads -> oi[m][512] f16
// ---------------------------------------------------------------------------
__global__ __launch_bounds__(256)
void k_rope_gather(const _Float16* __restrict__ O, const float* __restrict__ rot,
                   _Float16* __restrict__ oi) {
  int m = blockIdx.x;
  int b = m >> 12, n = m & 4095;
  int t = threadIdx.x;
  int h = t >> 5, pr = t & 31;
  int d0 = pr * 2, d1 = d0 + 1;
  const float* f = rot + (size_t)m * DH;
  float f0 = f[d0], f1 = f[d1];
  float c0 = __cosf(f0), s0 = __sinf(f0);
  float c1 = __cosf(f1), s1 = __sinf(f1);
  int bh = b * HEADS + h;
  size_t o = ((size_t)bh * SEQ + n) * DH;
  float x0 = (float)O[o + d0], x1 = (float)O[o + d1];
  float y0 = x0 * c0 + x1 * s0;   // rope with -freqs
  float y1 = x1 * c1 - x0 * s1;
  oi[(size_t)m * INNER + h * DH + d0] = (_Float16)y0;
  oi[(size_t)m * INNER + h * DH + d1] = (_Float16)y1;
}

// ---------------------------------------------------------------------------
extern "C" void kernel_launch(void* const* d_in, const int* in_sizes, int n_in,
                              void* d_out, int out_size, void* d_ws, size_t ws_size,
                              hipStream_t stream) {
  const float* x     = (const float*)d_in[0];
  const float* rot   = (const float*)d_in[1];
  const float* ln_w  = (const float*)d_in[2];
  const float* ln_b  = (const float*)d_in[3];
  const float* w_qkv = (const float*)d_in[4];
  const float* w_out = (const float*)d_in[5];
  const float* b_out = (const float*)d_in[6];
  float* out = (float*)d_out;

  char* ws = (char*)d_ws;
  _Float16* wqkv16 = (_Float16*)(ws);                    // 1536*512*2 = 1.5MB
  _Float16* wout16 = (_Float16*)(ws + 1572864);          // 512*512*2  = 0.5MB
  _Float16* xn16   = (_Float16*)(ws + 2097152);          // 8192*512*2 = 8MB
  _Float16* qkv16  = (_Float16*)(ws + 10485760);         // 8192*1536*2 = 24MB
  _Float16* qh     = (_Float16*)(ws + 35651584);         // 8MB
  _Float16* kh     = (_Float16*)(ws + 44040192);         // 8MB
  _Float16* vT     = (_Float16*)(ws + 52428800);         // 8MB -> total ~60.8MB
  _Float16* O16    = qkv16;   // reuse (qkv dead after scatter)
  _Float16* oi16   = xn16;    // reuse (xn dead after qkv gemm)

  k_cvt<<<(QKV_N * DIM + 255) / 256, 256, 0, stream>>>(w_qkv, wqkv16, QKV_N * DIM);
  k_cvt<<<(DIM * INNER + 255) / 256, 256, 0, stream>>>(w_out, wout16, DIM * INNER);
  k_ln<<<ROWS, 256, 0, stream>>>(x, ln_w, ln_b, xn16);

  dim3 g1(ROWS / 32, QKV_N / 256);
  k_gemm<<<g1, 256, 0, stream>>>(xn16, wqkv16, DIM, qkv16, nullptr, nullptr, QKV_N);

  k_rope_scatter<<<ROWS, 256, 0, stream>>>(qkv16, rot, qh, kh, vT);

  dim3 g2(SEQ / 128, BATCH * HEADS);
  k_attn<<<g2, 256, 0, stream>>>(qh, kh, vT, O16);

  k_rope_gather<<<ROWS, 256, 0, stream>>>(O16, rot, oi16);

  dim3 g3(ROWS / 32, DIM / 256);
  k_gemm<<<g3, 256, 0, stream>>>(oi16, wout16, INNER, nullptr, out, b_out, DIM);
}
